// DeformConv2D_5463198401088
// MI455X (gfx1250) — compile-verified, matrix operand order checked by reading, not run on hardware
//
#include <hip/hip_runtime.h>

// ---------------------------------------------------------------------------
// Deformable conv (b=8, C=64, 64x64, 3x3) on MI455X / gfx1250.
//
// Phase A (pack_x):  NCHW f32 -> zero-padded NHWC f16  (8 x 66 x 66 x 64)
// Phase B (pack_w):  weight (co,ci,kx,ky) f32 -> Wt[co][n*64+ci] f16
// Phase C (deform_wmma): bilinear gather straight into WMMA A-fragments,
//                        GEMM via v_wmma_f32_16x16x32_f16 (K=576 -> 18 steps).
//
// Fragment layouts (CDNA5 ISA 7.12.2, 16-bit A 16x32):
//   lane l, half = l>>4: element e (0..15) holds K = (e&7) + 8*half + 16*(e>>3)
//   => two contiguous 8-element K runs -> two b128 loads per operand fragment.
// B mirrors A with lane -> N column; C/D: lane%16 = N, vgpr r = M row
//   (+8 for lanes 16..31).
// ---------------------------------------------------------------------------

typedef _Float16 h8  __attribute__((ext_vector_type(8)));
typedef _Float16 h16 __attribute__((ext_vector_type(16)));
typedef float    f8  __attribute__((ext_vector_type(8)));

#define HP    66                 // padded spatial size (64 + 2*PAD)
#define KTOT  576                // 9 * 64
#define XH_ELEMS (8 * HP * HP * 64)
#define WT_ELEMS (64 * KTOT)

// ---- pre-pass 1: NCHW f32 -> padded NHWC f16 ------------------------------
__global__ void pack_x(const float* __restrict__ x, _Float16* __restrict__ xh) {
    int idx = blockIdx.x * 256 + threadIdx.x;
    if (idx >= XH_ELEMS) return;
    int c  = idx & 63;
    int t  = idx >> 6;
    int wp = t % HP; t /= HP;
    int hp = t % HP;
    int b  = t / HP;
    float v = 0.f;
    if (hp >= 1 && hp <= 64 && wp >= 1 && wp <= 64)
        v = x[((b * 64 + c) * 64 + (hp - 1)) * 64 + (wp - 1)];
    xh[idx] = (_Float16)v;
}

// ---- pre-pass 2: weight repack: Wt[co][n*64+ci] = W[co][ci][n] ------------
__global__ void pack_w(const float* __restrict__ w, _Float16* __restrict__ wt) {
    int idx = blockIdx.x * 256 + threadIdx.x;
    if (idx >= WT_ELEMS) return;
    int k  = idx % KTOT;
    int co = idx / KTOT;
    int n  = k >> 6;
    int ci = k & 63;
    wt[idx] = (_Float16)w[(co * 64 + ci) * 9 + n];
}

// ---- main kernel ----------------------------------------------------------
__global__ void __launch_bounds__(256)
deform_wmma(const float* __restrict__ offset, const _Float16* __restrict__ xh,
            const _Float16* __restrict__ wt, float* __restrict__ out) {
    extern __shared__ _Float16 w_lds[];            // 64 * 576 f16 = 72 KB

    // cooperative 128-bit copy of the repacked weights into LDS
    for (int i = threadIdx.x; i < WT_ELEMS / 8; i += 256)
        ((h8*)w_lds)[i] = ((const h8*)wt)[i];
    __syncthreads();

    const int lane = threadIdx.x & 31;
    const int wave = threadIdx.x >> 5;
    const int row  = lane & 15;                    // A: M row; B: N col
    const int lh   = lane >> 4;                    // lane half
    const int pix0 = blockIdx.x * 128 + wave * 16; // 16 consecutive pixels/wave
    const int pix  = pix0 + row;
    const int b    = pix >> 12;                    // pixels linear over (b,h,w)
    const int hw   = pix & 4095;
    const int h    = hw >> 6;
    const int w    = hw & 63;
    const int ci_off = 8 * lh;                     // lane's channel phase

    f8 acc[4] = {};                                // 4 co-tiles of 16

    for (int n = 0; n < 9; ++n) {
        // learned offsets: channel 2n = x(row) offset, 2n+1 = y(col) offset
        const float offx = offset[(b * 18 + 2 * n    ) * 4096 + hw];
        const float offy = offset[(b * 18 + 2 * n + 1) * 4096 + hw];
        const int dx = n / 3 - 1, dy = n % 3 - 1;

        float px = (float)(h + 1 + dx) + offx;     // padded coords
        float py = (float)(w + 1 + dy) + offy;
        float fpx = floorf(px), fpy = floorf(py);
        float qltx = fminf(fmaxf(fpx,        0.f), 65.f);
        float qlty = fminf(fmaxf(fpy,        0.f), 65.f);
        float qrbx = fminf(fmaxf(fpx + 1.f,  0.f), 65.f);
        float qrby = fminf(fmaxf(fpy + 1.f,  0.f), 65.f);
        bool  mx = (px < 1.f) || (px > 64.f);      // snap-to-floor mask
        bool  my = (py < 1.f) || (py > 64.f);
        float pxc = fminf(fmaxf(mx ? fpx : px, 0.f), 65.f);
        float pyc = fminf(fmaxf(my ? fpy : py, 0.f), 65.f);

        float glt = (1.f + (qltx - pxc)) * (1.f + (qlty - pyc));
        float grb = (1.f - (qrbx - pxc)) * (1.f - (qrby - pyc));
        float glb = (1.f + (qltx - pxc)) * (1.f - (qrby - pyc));
        float grt = (1.f - (qrbx - pxc)) * (1.f + (qlty - pyc));

        const int ix0 = (int)qltx, iy0 = (int)qlty;
        const int ix1 = (int)qrbx, iy1 = (int)qrby;
        // NHWC: each corner's 64 channels are contiguous f16
        const _Float16* blt = xh + (((b * HP + ix0) * HP + iy0) << 6) + ci_off;
        const _Float16* brb = xh + (((b * HP + ix1) * HP + iy1) << 6) + ci_off;
        const _Float16* blb = xh + (((b * HP + ix0) * HP + iy1) << 6) + ci_off;
        const _Float16* brt = xh + (((b * HP + ix1) * HP + iy0) << 6) + ci_off;
        const _Float16 hlt = (_Float16)glt, hrb = (_Float16)grb;
        const _Float16 hlb = (_Float16)glb, hrt = (_Float16)grt;

        #pragma unroll
        for (int s = 0; s < 2; ++s) {              // two K=32 steps per point
            const int o0 = 32 * s, o1 = 32 * s + 16;
            // bilinear blend directly in fragment order (packed f16 FMA)
            h8 a0 = *(const h8*)(blt + o0) * hlt + *(const h8*)(brb + o0) * hrb
                  + *(const h8*)(blb + o0) * hlb + *(const h8*)(brt + o0) * hrt;
            h8 a1 = *(const h8*)(blt + o1) * hlt + *(const h8*)(brb + o1) * hrb
                  + *(const h8*)(blb + o1) * hlb + *(const h8*)(brt + o1) * hrt;
            h16 afrag = __builtin_shufflevector(a0, a1, 0, 1, 2, 3, 4, 5, 6, 7,
                                                8, 9, 10, 11, 12, 13, 14, 15);
            const int kb = n * 64 + 32 * s + ci_off;
            #pragma unroll
            for (int t = 0; t < 4; ++t) {          // 4 co-tiles
                const _Float16* wp = w_lds + (t * 16 + row) * KTOT + kb;
                h8 b0 = *(const h8*)wp;
                h8 b1 = *(const h8*)(wp + 16);
                h16 bfrag = __builtin_shufflevector(b0, b1, 0, 1, 2, 3, 4, 5, 6,
                                                    7, 8, 9, 10, 11, 12, 13, 14,
                                                    15);
                acc[t] = __builtin_amdgcn_wmma_f32_16x16x32_f16(
                    false, afrag, false, bfrag, (short)0, acc[t], false, false);
            }
        }
    }

    // C/D layout: lane%16 = N (co), vgpr r = M row (+8 for upper lane half)
    #pragma unroll
    for (int t = 0; t < 4; ++t) {
        const int co = t * 16 + row;
        #pragma unroll
        for (int r = 0; r < 8; ++r) {
            const int m   = r + 8 * lh;
            const int p   = pix0 + m;
            const int bb  = p >> 12;
            const int phw = p & 4095;
            out[(bb * 64 + co) * 4096 + phw] = acc[t][r];
        }
    }
}

// ---------------------------------------------------------------------------
extern "C" void kernel_launch(void* const* d_in, const int* in_sizes, int n_in,
                              void* d_out, int out_size, void* d_ws, size_t ws_size,
                              hipStream_t stream) {
    const float* x      = (const float*)d_in[0];
    const float* offset = (const float*)d_in[1];
    const float* weight = (const float*)d_in[2];
    float* out = (float*)d_out;

    _Float16* xh = (_Float16*)d_ws;                          // 4,460,544 B
    _Float16* wt = (_Float16*)((char*)d_ws + (size_t)XH_ELEMS * 2); // 73,728 B

    pack_x<<<(XH_ELEMS + 255) / 256, 256, 0, stream>>>(x, xh);
    pack_w<<<(WT_ELEMS + 255) / 256, 256, 0, stream>>>(weight, wt);

    // 32768 pixels / (8 waves * 16 pixels) = 256 blocks; 72 KB dynamic LDS
    deform_wmma<<<256, 256, WT_ELEMS * sizeof(_Float16), stream>>>(
        offset, xh, wt, out);
}
